// MS_Block_Spike_SepConv_15633680957883
// MI455X (gfx1250) — matrix operand order repaired and checked by reading, not compiled, not run on hardware
//
#include <hip/hip_runtime.h>

// ---------------------------------------------------------------------------
// MS_Block (spiking sepconv + linear attention + MLP) for gfx1250.
// All large GEMMs run on v_wmma_f32_16x16x32_f16 with the LIF (T=4) scan fused
// into the GEMM epilogue and frozen-BN folded into f16 weights.
// Spike tensors are stored in the layout their consumer wants:
//   n-major  [tb][n][c]  -> WMMA B operand loads are contiguous 32B per lane
//   c-major  [tb][c][n]  -> used only by depthwise conv and the kv (K=n) GEMM
// so no transposes / LDS staging are needed anywhere. Prefetches use high
// locality so they lower to WGP-scope (fill near caches, not just GL2).
// ---------------------------------------------------------------------------

typedef _Float16 half_t;
typedef __attribute__((ext_vector_type(16))) _Float16 v16h;
typedef __attribute__((ext_vector_type(8)))  _Float16 v8h;
typedef __attribute__((ext_vector_type(8)))  float    v8f;

#define T_N     4
#define B_N     16
#define TB_N    64
#define C_N     256
#define MED_N   512
#define CV_N    1024
#define HID_N   1024
#define N_N     576          // 24*24
#define W_IMG   24
#define HEADS_N 8
#define DH_N    32           // q/k head dim
#define DHV_N   128          // v head dim (CV/HEADS)
#define LIF_DECAY 0.25f
#define LIF_TH    0.5f
#define BN_EPS    1e-5f

#define WMMA_F16(a, b, c) \
  __builtin_amdgcn_wmma_f32_16x16x32_f16(false, (a), false, (b), (short)0, (c), false, false)

#define PREFETCH_NEAR(p) __builtin_prefetch((p), 0, 3)  // high locality -> WGP-scope

static __device__ __forceinline__ v16h make_v16(v8h lo, v8h hi) {
  v16h r;
#pragma unroll
  for (int i = 0; i < 8; ++i) { r[i] = lo[i]; r[8 + i] = hi[i]; }
  return r;
}

// A fragment (16x32 f16, MxK): 'row' holds 32 contiguous K halfs.
// element e -> K = khalf*16 + e (contiguous halfs)
static __device__ __forceinline__ v16h load_a_frag(const half_t* row, int khalf) {
  v8h lo = *(const v8h*)(row + khalf * 16);
  v8h hi = *(const v8h*)(row + khalf * 16 + 8);
  return make_v16(lo, hi);
}

// B fragment (32x16 f16, KxN) from n-major storage: 'col' points at the K run
// for this lane's column; element e -> K = khalf*16 + e (contiguous halfs).
static __device__ __forceinline__ v16h load_b_frag_nmajor(const half_t* col, int khalf) {
  v8h lo = *(const v8h*)(col + khalf * 16);
  v8h hi = *(const v8h*)(col + khalf * 16 + 8);
  return make_v16(lo, hi);
}

// ---------------------------------------------------------------------------
// Prep kernels: fold frozen BN (+ optional fc bias) into f16 weights / f32 bias
// ---------------------------------------------------------------------------
__global__ void prep_weight_f16(const float* __restrict__ W, const float* __restrict__ gamma,
                                const float* __restrict__ var, half_t* __restrict__ Wo,
                                int M, int K) {
  int i = blockIdx.x * 256 + threadIdx.x;
  if (i >= M * K) return;
  int m = i / K;
  float sc = gamma[m] * rsqrtf(var[m] + BN_EPS);
  Wo[i] = (half_t)(W[i] * sc);
}

__global__ void prep_bias_f32(const float* __restrict__ gamma, const float* __restrict__ beta,
                              const float* __restrict__ mean, const float* __restrict__ var,
                              const float* __restrict__ fcb, float* __restrict__ bo, int M) {
  int m = blockIdx.x * 256 + threadIdx.x;
  if (m >= M) return;
  float sc = gamma[m] * rsqrtf(var[m] + BN_EPS);
  float b0 = fcb ? fcb[m] : 0.f;
  bo[m] = beta[m] + sc * (b0 - mean[m]);
}

__global__ void prep_dw_f32(const float* __restrict__ wd, const float* __restrict__ gamma,
                            const float* __restrict__ var, float* __restrict__ Wo) {
  int i = blockIdx.x * 256 + threadIdx.x;
  if (i >= MED_N * 9) return;
  int c = i / 9;
  Wo[i] = wd[i] * gamma[c] * rsqrtf(var[c] + BN_EPS);
}

// ---------------------------------------------------------------------------
// LIF over T on an f32 tensor [T,B,Cc,N] (optionally + positional embedding),
// emitting f16 spikes in n-major layout [tb][n][c].
// ---------------------------------------------------------------------------
__global__ void lif_spike_kernel(const float* __restrict__ X, half_t* __restrict__ S,
                                 const float* __restrict__ s_pos, const float* __restrict__ t_pos,
                                 const int* __restrict__ branch, int Cc, int usePos) {
  size_t idx = (size_t)blockIdx.x * 256 + threadIdx.x;
  size_t tot = (size_t)B_N * Cc * N_N;
  if (idx >= tot) return;
  int cn = (int)(idx % ((size_t)Cc * N_N));
  int b = (int)(idx / ((size_t)Cc * N_N));
  int c = cn / N_N, n = cn % N_N;
  int br = usePos ? branch[0] : 0;
  float mem = 0.f;
#pragma unroll
  for (int t = 0; t < T_N; ++t) {
    float xv = X[(size_t)t * tot + idx];
    if (usePos)
      xv += (br == 0) ? s_pos[(size_t)c * N_N + n]
                      : t_pos[((size_t)t * Cc + c) * N_N + n];  // idx=min(t,3)=t for T=4
    mem = mem * LIF_DECAY + xv;
    float s = mem > LIF_TH ? 1.f : 0.f;
    S[((size_t)(t * B_N + b) * N_N + n) * Cc + c] = (half_t)s;
    mem *= (1.f - s);
  }
}

// ---------------------------------------------------------------------------
// Depthwise 3x3 on c-major spike input (BN folded), fused LIF -> n-major spikes
// ---------------------------------------------------------------------------
__global__ void dw_lif_kernel(const half_t* __restrict__ S2, const float* __restrict__ Wd,
                              const float* __restrict__ bd, half_t* __restrict__ S3) {
  size_t idx = (size_t)blockIdx.x * 256 + threadIdx.x;
  size_t tot = (size_t)B_N * MED_N * N_N;
  if (idx >= tot) return;
  int n = (int)(idx % N_N);
  int bc = (int)(idx / N_N);
  int c = bc % MED_N, b = bc / MED_N;
  int hh = n / W_IMG, ww = n % W_IMG;
  float wl[9];
#pragma unroll
  for (int i = 0; i < 9; ++i) wl[i] = Wd[c * 9 + i];
  float bias = bd[c];
  float mem = 0.f;
#pragma unroll
  for (int t = 0; t < T_N; ++t) {
    size_t base = ((size_t)(t * B_N + b) * MED_N + c) * N_N;
    float z = bias;
#pragma unroll
    for (int ky = -1; ky <= 1; ++ky) {
      int yy = hh + ky;
      if (yy < 0 || yy >= W_IMG) continue;
#pragma unroll
      for (int kx = -1; kx <= 1; ++kx) {
        int xx = ww + kx;
        if (xx < 0 || xx >= W_IMG) continue;
        z += wl[(ky + 1) * 3 + (kx + 1)] * (float)S2[base + yy * W_IMG + xx];
      }
    }
    mem = mem * LIF_DECAY + z;
    float s = mem > LIF_TH ? 1.f : 0.f;
    S3[((size_t)(t * B_N + b) * N_N + n) * MED_N + c] = (half_t)s;
    mem *= (1.f - s);
  }
}

// ---------------------------------------------------------------------------
// Fused GEMM + LIF. A = W[M,K] f16 (BN folded), B = spikes n-major [tb][n][K].
// One block: 64(M) x 32(N) tile for one b, all 4 time steps (A shared).
// OUT_NMAJOR=1 -> spikes [tb][n][M] (16B vector stores); 0 -> [tb][M][n].
// grid = (M/64, N/32, B), block = 256 (8 waves: 4x2 of 16x16 tiles).
// ---------------------------------------------------------------------------
template <int OUT_NMAJOR>
__global__ __launch_bounds__(256) void gemm_lif_kernel(
    const half_t* __restrict__ W, const float* __restrict__ bias,
    const half_t* __restrict__ S, half_t* __restrict__ Out, int M, int K) {
  int tid = threadIdx.x;
  int b = blockIdx.z, m0 = blockIdx.x * 64, n0 = blockIdx.y * 32;
  int lane = tid & 31, wv = tid >> 5;
  int m_w = (wv >> 1) * 16, n_w = (wv & 1) * 16;
  int khalf = lane >> 4, l16 = lane & 15;
  int n = n0 + n_w + l16;

  v8f zero = {};
  v8f acc[T_N];
#pragma unroll
  for (int t = 0; t < T_N; ++t) acc[t] = zero;

  const half_t* arow = W + (size_t)(m0 + m_w + l16) * K;
  const half_t* brow[T_N];
#pragma unroll
  for (int t = 0; t < T_N; ++t)
    brow[t] = S + ((size_t)(t * B_N + b) * N_N + n) * K;

  for (int k0 = 0; k0 < K; k0 += 32) {
    PREFETCH_NEAR(arow + k0 + 32);
    PREFETCH_NEAR(brow[0] + k0 + 32);
    v16h a = load_a_frag(arow + k0, khalf);
#pragma unroll
    for (int t = 0; t < T_N; ++t) {
      v16h bf = load_b_frag_nmajor(brow[t] + k0, khalf);
      acc[t] = WMMA_F16(a, bf, acc[t]);
    }
  }

  int mbase = m0 + m_w + khalf * 8;
  float bs[8], mem[8];
#pragma unroll
  for (int j = 0; j < 8; ++j) { bs[j] = bias[mbase + j]; mem[j] = 0.f; }
#pragma unroll
  for (int t = 0; t < T_N; ++t) {
    v8h sp;
#pragma unroll
    for (int j = 0; j < 8; ++j) {
      mem[j] = mem[j] * LIF_DECAY + acc[t][j] + bs[j];
      float s = mem[j] > LIF_TH ? 1.f : 0.f;
      sp[j] = (half_t)s;
      mem[j] *= (1.f - s);
    }
    if (OUT_NMAJOR) {
      *(v8h*)(Out + ((size_t)(t * B_N + b) * N_N + n) * M + mbase) = sp;
    } else {
#pragma unroll
      for (int j = 0; j < 8; ++j)
        Out[((size_t)(t * B_N + b) * M + mbase + j) * N_N + n] = sp[j];
    }
  }
}

// Same GEMM body, epilogue = residual add into f32 [T,B,M,N] output (no LIF).
__global__ __launch_bounds__(256) void gemm_add_kernel(
    const half_t* __restrict__ W, const float* __restrict__ bias,
    const half_t* __restrict__ S, const float* Res, float* Out, int M, int K) {
  int tid = threadIdx.x;
  int b = blockIdx.z, m0 = blockIdx.x * 64, n0 = blockIdx.y * 32;
  int lane = tid & 31, wv = tid >> 5;
  int m_w = (wv >> 1) * 16, n_w = (wv & 1) * 16;
  int khalf = lane >> 4, l16 = lane & 15;
  int n = n0 + n_w + l16;

  v8f zero = {};
  v8f acc[T_N];
#pragma unroll
  for (int t = 0; t < T_N; ++t) acc[t] = zero;

  const half_t* arow = W + (size_t)(m0 + m_w + l16) * K;
  const half_t* brow[T_N];
#pragma unroll
  for (int t = 0; t < T_N; ++t)
    brow[t] = S + ((size_t)(t * B_N + b) * N_N + n) * K;

  for (int k0 = 0; k0 < K; k0 += 32) {
    PREFETCH_NEAR(arow + k0 + 32);
    PREFETCH_NEAR(brow[0] + k0 + 32);
    v16h a = load_a_frag(arow + k0, khalf);
#pragma unroll
    for (int t = 0; t < T_N; ++t) {
      v16h bf = load_b_frag_nmajor(brow[t] + k0, khalf);
      acc[t] = WMMA_F16(a, bf, acc[t]);
    }
  }

  int mbase = m0 + m_w + khalf * 8;
#pragma unroll
  for (int j = 0; j < 8; ++j) {
    float bsj = bias[mbase + j];
#pragma unroll
    for (int t = 0; t < T_N; ++t) {
      size_t o = ((size_t)(t * B_N + b) * M + mbase + j) * N_N + n;
      Out[o] = Res[o] + acc[t][j] + bsj;
    }
  }
}

// ---------------------------------------------------------------------------
// kvT[tb,h,e,d] = sum_n Sv[tb, h*128+e, n] * Sk[tb, h*32+d, n]   (f16 out)
// Sk/Sv are c-major, so A rows and B K-runs are contiguous along n.
// One wave per (tb, h, e-quad of 32): 2x2 WMMA tiles, K loop over n (576).
// ---------------------------------------------------------------------------
__global__ __launch_bounds__(256) void kv_kernel(const half_t* __restrict__ Sk,
                                                 const half_t* __restrict__ Sv,
                                                 half_t* __restrict__ kvT) {
  int tid = threadIdx.x;
  int lane = tid & 31, wv = tid >> 5;
  int gid = blockIdx.x * 8 + wv;          // 0 .. 2047
  int eq = gid & 3;
  int h = (gid >> 2) & 7;
  int tb = gid >> 5;
  int khalf = lane >> 4, l16 = lane & 15;

  v8f zero = {};
  v8f acc[2][2];
#pragma unroll
  for (int e2 = 0; e2 < 2; ++e2)
#pragma unroll
    for (int d2 = 0; d2 < 2; ++d2) acc[e2][d2] = zero;

  const half_t* vbase = Sv + ((size_t)tb * CV_N + h * DHV_N + eq * 32) * N_N;
  const half_t* kbase = Sk + ((size_t)tb * C_N + h * DH_N) * N_N;

  for (int n0 = 0; n0 < N_N; n0 += 32) {
    v16h af[2], bf[2];
#pragma unroll
    for (int e2 = 0; e2 < 2; ++e2)
      af[e2] = load_a_frag(vbase + (size_t)(e2 * 16 + l16) * N_N + n0, khalf);
#pragma unroll
    for (int d2 = 0; d2 < 2; ++d2) {
      const half_t* p = kbase + (size_t)(d2 * 16 + l16) * N_N + n0 + khalf * 16;
      bf[d2] = make_v16(*(const v8h*)p, *(const v8h*)(p + 8));
    }
#pragma unroll
    for (int e2 = 0; e2 < 2; ++e2)
#pragma unroll
      for (int d2 = 0; d2 < 2; ++d2) acc[e2][d2] = WMMA_F16(af[e2], bf[d2], acc[e2][d2]);
  }
  half_t* outp = kvT + (((size_t)tb * HEADS_N + h) * DHV_N + eq * 32) * DH_N;
#pragma unroll
  for (int e2 = 0; e2 < 2; ++e2)
#pragma unroll
    for (int d2 = 0; d2 < 2; ++d2)
#pragma unroll
      for (int j = 0; j < 8; ++j)
        outp[(e2 * 16 + khalf * 8 + j) * DH_N + d2 * 16 + l16] = (half_t)acc[e2][d2][j];
}

// ---------------------------------------------------------------------------
// o^T[e,n] = sum_d kvT[e,d] * Sq[d,n], scaled by 2*SCALE, fused LIF -> spikes
// So n-major [tb][n][h*128+e]. Sq is n-major so B loads are contiguous.
// grid=(N/16, HEADS, B), block=256: wave wv -> e tile wv*16 (128 channels).
// ---------------------------------------------------------------------------
__global__ __launch_bounds__(256) void attn_o_kernel(const half_t* __restrict__ kvT,
                                                     const half_t* __restrict__ Sq,
                                                     half_t* __restrict__ So) {
  const float SC2 = 0.35355339059327373f;  // 2 * (32)^-0.5
  int tid = threadIdx.x;
  int lane = tid & 31, wv = tid >> 5;
  int b = blockIdx.z, h = blockIdx.y;
  int khalf = lane >> 4, l16 = lane & 15;
  int n = blockIdx.x * 16 + l16;
  int e_w = wv * 16;

  v8f zero = {};
  v8f acc[T_N];
#pragma unroll
  for (int t = 0; t < T_N; ++t) {
    int tb = t * B_N + b;
    const half_t* arow = kvT + (((size_t)tb * HEADS_N + h) * DHV_N + e_w + l16) * DH_N;
    v16h a = load_a_frag(arow, khalf);
    const half_t* qcol = Sq + ((size_t)tb * N_N + n) * C_N + h * DH_N;
    v16h bf = load_b_frag_nmajor(qcol, khalf);
    acc[t] = WMMA_F16(a, bf, zero);
  }
  int ebase = e_w + khalf * 8;
  float mem[8];
#pragma unroll
  for (int j = 0; j < 8; ++j) mem[j] = 0.f;
#pragma unroll
  for (int t = 0; t < T_N; ++t) {
    v8h sp;
#pragma unroll
    for (int j = 0; j < 8; ++j) {
      mem[j] = mem[j] * LIF_DECAY + acc[t][j] * SC2;
      float s = mem[j] > LIF_TH ? 1.f : 0.f;
      sp[j] = (half_t)s;
      mem[j] *= (1.f - s);
    }
    *(v8h*)(So + ((size_t)(t * B_N + b) * N_N + n) * CV_N + h * DHV_N + ebase) = sp;
  }
}

// ---------------------------------------------------------------------------
// Host side
// ---------------------------------------------------------------------------
// d_in indices (setup_inputs dict insertion order, nested dicts in place):
// 0:x 1:sep_w1 2-5:sep_bn1(g,b,m,v) 6:sep_wd 7-10:sep_bnd 11:sep_w2
// 12-15:sep_bn2 16:wq 17-20:bnq 21:wk 22-25:bnk 26:wv 27-30:bnv 31:wp
// 32-35:bnp 36:s_pos 37:t_pos 38:fc1_w 39:fc1_b 40-43:fc1_bn 44:fc2_w
// 45:fc2_b 46-49:fc2_bn 50:branch

extern "C" void kernel_launch(void* const* d_in, const int* in_sizes, int n_in,
                              void* d_out, int out_size, void* d_ws, size_t ws_size,
                              hipStream_t stream) {
  (void)in_sizes; (void)n_in; (void)out_size; (void)ws_size;
  char* ws = (char*)d_ws;
  size_t off = 0;
  auto alloc = [&](size_t bytes) { size_t r = off; off += (bytes + 255) & ~(size_t)255; return r; };

  const size_t spA_o = alloc((size_t)TB_N * C_N * N_N * 2);    // S1 / Sx / Sm (n-major)
  const size_t spB_o = alloc((size_t)TB_N * CV_N * N_N * 2);   // S2(c-major) / Sv(c-major) / So / Sh1
  const size_t spC_o = alloc((size_t)TB_N * MED_N * N_N * 2);  // S3 (n-major)
  const size_t spQ_o = alloc((size_t)TB_N * C_N * N_N * 2);    // Sq (n-major)
  const size_t spK_o = alloc((size_t)TB_N * C_N * N_N * 2);    // Sk (c-major)
  const size_t kvt_o = alloc((size_t)TB_N * HEADS_N * DHV_N * DH_N * 2);
  const size_t w1_o = alloc(MED_N * C_N * 2);
  const size_t w2_o = alloc(C_N * MED_N * 2);
  const size_t wq_o = alloc(C_N * C_N * 2);
  const size_t wk_o = alloc(C_N * C_N * 2);
  const size_t wv_o = alloc(CV_N * C_N * 2);
  const size_t wp_o = alloc(C_N * CV_N * 2);
  const size_t wf1_o = alloc(HID_N * C_N * 2);
  const size_t wf2_o = alloc(C_N * HID_N * 2);
  const size_t wd_o = alloc(MED_N * 9 * 4);
  const size_t b1_o = alloc(MED_N * 4);
  const size_t b2_o = alloc(C_N * 4);
  const size_t bq_o = alloc(C_N * 4);
  const size_t bk_o = alloc(C_N * 4);
  const size_t bv_o = alloc(CV_N * 4);
  const size_t bp_o = alloc(C_N * 4);
  const size_t bf1_o = alloc(HID_N * 4);
  const size_t bf2_o = alloc(C_N * 4);
  const size_t bd_o = alloc(MED_N * 4);

  auto H = [&](size_t o) { return (half_t*)(ws + o); };
  auto Fw = [&](size_t o) { return (float*)(ws + o); };
  auto FI = [&](int i) { return (const float*)d_in[i]; };

  const float* x_in = FI(0);
  float* out = (float*)d_out;
  const int* branch = (const int*)d_in[50];

  // ---- weight/bias folding (tiny) ----
  auto gw = [](int n) { return dim3((n + 255) / 256); };
  prep_weight_f16<<<gw(MED_N * C_N), 256, 0, stream>>>(FI(1), FI(2), FI(5), H(w1_o), MED_N, C_N);
  prep_weight_f16<<<gw(C_N * MED_N), 256, 0, stream>>>(FI(11), FI(12), FI(15), H(w2_o), C_N, MED_N);
  prep_weight_f16<<<gw(C_N * C_N), 256, 0, stream>>>(FI(16), FI(17), FI(20), H(wq_o), C_N, C_N);
  prep_weight_f16<<<gw(C_N * C_N), 256, 0, stream>>>(FI(21), FI(22), FI(25), H(wk_o), C_N, C_N);
  prep_weight_f16<<<gw(CV_N * C_N), 256, 0, stream>>>(FI(26), FI(27), FI(30), H(wv_o), CV_N, C_N);
  prep_weight_f16<<<gw(C_N * CV_N), 256, 0, stream>>>(FI(31), FI(32), FI(35), H(wp_o), C_N, CV_N);
  prep_weight_f16<<<gw(HID_N * C_N), 256, 0, stream>>>(FI(38), FI(40), FI(43), H(wf1_o), HID_N, C_N);
  prep_weight_f16<<<gw(C_N * HID_N), 256, 0, stream>>>(FI(44), FI(46), FI(49), H(wf2_o), C_N, HID_N);
  prep_dw_f32<<<gw(MED_N * 9), 256, 0, stream>>>(FI(6), FI(7), FI(10), Fw(wd_o));
  prep_bias_f32<<<gw(MED_N), 256, 0, stream>>>(FI(2), FI(3), FI(4), FI(5), nullptr, Fw(b1_o), MED_N);
  prep_bias_f32<<<gw(MED_N), 256, 0, stream>>>(FI(7), FI(8), FI(9), FI(10), nullptr, Fw(bd_o), MED_N);
  prep_bias_f32<<<gw(C_N), 256, 0, stream>>>(FI(12), FI(13), FI(14), FI(15), nullptr, Fw(b2_o), C_N);
  prep_bias_f32<<<gw(C_N), 256, 0, stream>>>(FI(17), FI(18), FI(19), FI(20), nullptr, Fw(bq_o), C_N);
  prep_bias_f32<<<gw(C_N), 256, 0, stream>>>(FI(22), FI(23), FI(24), FI(25), nullptr, Fw(bk_o), C_N);
  prep_bias_f32<<<gw(CV_N), 256, 0, stream>>>(FI(27), FI(28), FI(29), FI(30), nullptr, Fw(bv_o), CV_N);
  prep_bias_f32<<<gw(C_N), 256, 0, stream>>>(FI(32), FI(33), FI(34), FI(35), nullptr, Fw(bp_o), C_N);
  prep_bias_f32<<<gw(HID_N), 256, 0, stream>>>(FI(40), FI(41), FI(42), FI(43), FI(39), Fw(bf1_o), HID_N);
  prep_bias_f32<<<gw(C_N), 256, 0, stream>>>(FI(46), FI(47), FI(48), FI(49), FI(45), Fw(bf2_o), C_N);

  const int lifBlocksC = (B_N * C_N * N_N + 255) / 256;

  // ---- stage 1: sepconv ----
  lif_spike_kernel<<<lifBlocksC, 256, 0, stream>>>(x_in, H(spA_o), nullptr, nullptr, nullptr, C_N, 0);
  gemm_lif_kernel<0><<<dim3(MED_N / 64, N_N / 32, B_N), 256, 0, stream>>>(
      H(w1_o), Fw(b1_o), H(spA_o), H(spB_o), MED_N, C_N);            // S2 c-major (for dw)
  dw_lif_kernel<<<(B_N * MED_N * N_N + 255) / 256, 256, 0, stream>>>(
      H(spB_o), Fw(wd_o), Fw(bd_o), H(spC_o));                       // S3 n-major
  gemm_add_kernel<<<dim3(C_N / 64, N_N / 32, B_N), 256, 0, stream>>>(
      H(w2_o), Fw(b2_o), H(spC_o), x_in, out, C_N, MED_N);

  // ---- stage 2: linear attention ----
  lif_spike_kernel<<<lifBlocksC, 256, 0, stream>>>(out, H(spA_o), FI(36), FI(37), branch, C_N, 1);
  gemm_lif_kernel<1><<<dim3(C_N / 64, N_N / 32, B_N), 256, 0, stream>>>(
      H(wq_o), Fw(bq_o), H(spA_o), H(spQ_o), C_N, C_N);              // Sq n-major
  gemm_lif_kernel<0><<<dim3(C_N / 64, N_N / 32, B_N), 256, 0, stream>>>(
      H(wk_o), Fw(bk_o), H(spA_o), H(spK_o), C_N, C_N);              // Sk c-major
  gemm_lif_kernel<0><<<dim3(CV_N / 64, N_N / 32, B_N), 256, 0, stream>>>(
      H(wv_o), Fw(bv_o), H(spA_o), H(spB_o), CV_N, C_N);             // Sv c-major
  kv_kernel<<<dim3(TB_N * HEADS_N * 4 / 8), 256, 0, stream>>>(H(spK_o), H(spB_o), H(kvt_o));
  attn_o_kernel<<<dim3(N_N / 16, HEADS_N, B_N), 256, 0, stream>>>(H(kvt_o), H(spQ_o), H(spB_o));
  gemm_add_kernel<<<dim3(C_N / 64, N_N / 32, B_N), 256, 0, stream>>>(
      H(wp_o), Fw(bp_o), H(spB_o), out, out, C_N, CV_N);

  // ---- stage 3: MLP ----
  lif_spike_kernel<<<lifBlocksC, 256, 0, stream>>>(out, H(spA_o), nullptr, nullptr, nullptr, C_N, 0);
  gemm_lif_kernel<1><<<dim3(HID_N / 64, N_N / 32, B_N), 256, 0, stream>>>(
      H(wf1_o), Fw(bf1_o), H(spA_o), H(spB_o), HID_N, C_N);          // Sh1 n-major
  gemm_add_kernel<<<dim3(C_N / 64, N_N / 32, B_N), 256, 0, stream>>>(
      H(wf2_o), Fw(bf2_o), H(spB_o), out, out, C_N, HID_N);
}
